// TemporalTransformerBlock_35562329211322
// MI455X (gfx1250) — matrix operand order repaired and checked by reading, not compiled, not run on hardware
//
#include <hip/hip_runtime.h>
#include <hip/hip_bf16.h>
#include <math.h>

typedef __attribute__((ext_vector_type(16))) _Float16 v16h;
typedef __attribute__((ext_vector_type(8)))  _Float16 v8h;
typedef __attribute__((ext_vector_type(8)))  float    v8f;
typedef __attribute__((ext_vector_type(4)))  int      v4i;

#define AS1 __attribute__((address_space(1)))
#define AS3 __attribute__((address_space(3)))

// ---- problem constants ----
constexpr int BATCH = 4;
constexpr int SEQ   = 2048;
constexpr int EMBED = 512;
constexpr int HEADS = 8;
constexpr int HDIM  = 64;
constexpr int H3    = 3 * EMBED;     // 1536
constexpr int MLPH  = 4 * EMBED;     // 2048
constexpr int TOK   = BATCH * SEQ;   // 8192

// ----------------------------------------------------------------------------
// fragment helpers (wave32, v_wmma_f32_16x16x32_f16; contiguous K order used here, not AMD's table order)
// A (16x32): lanes 0-15 row M=l, K 0-15 ; lanes 16-31 row M=l, K 16-31
// B (32x16): lanes 0-15 col N=l, K 0-15 ; lanes 16-31 col N=l, K 16-31
// ----------------------------------------------------------------------------
__device__ __forceinline__ v16h frag_a(const _Float16* base) {
  v8h lo = *reinterpret_cast<const v8h*>(base);
  v8h hi = *reinterpret_cast<const v8h*>(base + 8);
  return __builtin_shufflevector(lo, hi, 0,1,2,3,4,5,6,7,8,9,10,11,12,13,14,15);
}
__device__ __forceinline__ v16h frag_b(const _Float16* base) {
  v8h lo = *reinterpret_cast<const v8h*>(base);
  v8h hi = *reinterpret_cast<const v8h*>(base + 8);
  return __builtin_shufflevector(lo, hi, 0,1,2,3,4,5,6,7,8,9,10,11,12,13,14,15);
}
__device__ __forceinline__ v8f wmma(v16h a, v16h b, v8f c) {
  return __builtin_amdgcn_wmma_f32_16x16x32_f16(false, a, false, b, (short)0, c, false, false);
}

// ----------------------------------------------------------------------------
// 16-lane-row butterfly reductions via v_permlane16_b32 (pure VALU; lanes
// 16-31 form an independent row, matching the WMMA C-layout halves).
// ----------------------------------------------------------------------------
#if __has_builtin(__builtin_amdgcn_permlane16)
#define USE_PERMLANE 1
__device__ __forceinline__ float pl16(float v, unsigned s0, unsigned s1) {
  unsigned u = __builtin_bit_cast(unsigned, v);
  unsigned r = __builtin_amdgcn_permlane16(u, u, s0, s1, false, false);
  return __builtin_bit_cast(float, r);
}
#endif
__device__ __forceinline__ float rowmax16(float x) {
#if defined(USE_PERMLANE)
  x = fmaxf(x, pl16(x, 0x67452301u, 0xEFCDAB89u));  // xor 1
  x = fmaxf(x, pl16(x, 0x54761032u, 0xDCFE98BAu));  // xor 2
  x = fmaxf(x, pl16(x, 0x32107654u, 0xBA98FEDCu));  // xor 4
  x = fmaxf(x, pl16(x, 0xFEDCBA98u, 0x76543210u));  // xor 8
#else
#pragma unroll
  for (int off = 1; off < 16; off <<= 1) x = fmaxf(x, __shfl_xor(x, off, 32));
#endif
  return x;
}
__device__ __forceinline__ float rowsum16(float x) {
#if defined(USE_PERMLANE)
  x += pl16(x, 0x67452301u, 0xEFCDAB89u);
  x += pl16(x, 0x54761032u, 0xDCFE98BAu);
  x += pl16(x, 0x32107654u, 0xBA98FEDCu);
  x += pl16(x, 0xFEDCBA98u, 0x76543210u);
#else
#pragma unroll
  for (int off = 1; off < 16; off <<= 1) x += __shfl_xor(x, off, 32);
#endif
  return x;
}

// ----------------------------------------------------------------------------
// Async global->LDS copy (CDNA5 GLOBAL_LOAD_ASYNC_TO_LDS_B128), guarded.
// Probe-revealed signature: (addrspace(1) int4*, addrspace(3) int4*, int, int)
// ----------------------------------------------------------------------------
#if __has_builtin(__builtin_amdgcn_global_load_async_to_lds_b128)
#define HAVE_ASYNC_COPY 1
#else
#define HAVE_ASYNC_COPY 0
#endif

__device__ __forceinline__ void copy16_g2l(const _Float16* gp, _Float16* lp) {
#if HAVE_ASYNC_COPY
  __builtin_amdgcn_global_load_async_to_lds_b128(
      (AS1 v4i*)gp, (AS3 v4i*)lp, 0, 0);
#else
  *reinterpret_cast<v8h*>(lp) = *reinterpret_cast<const v8h*>(gp);
#endif
}
__device__ __forceinline__ void async_wait0() {
#if HAVE_ASYNC_COPY
#if __has_builtin(__builtin_amdgcn_s_wait_asynccnt)
  __builtin_amdgcn_s_wait_asynccnt(0);
#else
  asm volatile("s_wait_asynccnt 0x0" ::: "memory");
#endif
#endif
}

// ---------------------------------------------------------------------------
// weight transpose + f32->f16 convert:  in[K][N] row-major -> out[N][K]
// ---------------------------------------------------------------------------
__global__ void k_transpose_cvt(const float* __restrict__ in, _Float16* __restrict__ out,
                                int K, int N) {
  int idx = blockIdx.x * 256 + threadIdx.x;
  if (idx >= K * N) return;
  int k = idx / N, n = idx % N;
  out[(size_t)n * K + k] = (_Float16)in[idx];
}

// ---------------------------------------------------------------------------
// LayerNorm (f32 in) -> f16 out.  One wave per 512-wide row, 8 rows / block.
// ---------------------------------------------------------------------------
__global__ void k_layernorm_f16(const float* __restrict__ x, const float* __restrict__ g,
                                const float* __restrict__ b, _Float16* __restrict__ out) {
  const int wave = threadIdx.x >> 5;
  const int lane = threadIdx.x & 31;
  const int row  = blockIdx.x * 8 + wave;
  const float* xr = x + (size_t)row * EMBED;

  float v[16];
  float s = 0.f;
#pragma unroll
  for (int i = 0; i < 16; ++i) { v[i] = xr[lane + 32 * i]; s += v[i]; }
#pragma unroll
  for (int off = 1; off < 32; off <<= 1) s += __shfl_xor(s, off, 32);
  const float mu = s * (1.0f / EMBED);

  float vs = 0.f;
#pragma unroll
  for (int i = 0; i < 16; ++i) { float d = v[i] - mu; vs += d * d; }
#pragma unroll
  for (int off = 1; off < 32; off <<= 1) vs += __shfl_xor(vs, off, 32);
  const float rs = rsqrtf(vs * (1.0f / EMBED) + 1e-5f);

  _Float16* orow = out + (size_t)row * EMBED;
#pragma unroll
  for (int i = 0; i < 16; ++i) {
    int e = lane + 32 * i;
    orow[e] = (_Float16)((v[i] - mu) * rs * g[e] + b[e]);
  }
}

// ---------------------------------------------------------------------------
// WMMA GEMM:  D[M][N] = A[M][K] (f16, row-major) * BT[N][K] (f16, N-major) + bias
//   EPI 0: f16 out  | EPI 1: f16 out + GELU | EPI 2: f32 out + residual
// block 256 (8 waves); tile 128x128; wave -> 64x32 (4x2 subtiles); kstep 32.
// A tile double-buffered through LDS with async DMA overlapping the WMMAs;
// B fragments direct from L2 with next-step prefetch.
// ---------------------------------------------------------------------------
template <int EPI>
__global__ __launch_bounds__(256) void k_wmma_gemm(
    const _Float16* __restrict__ A, const _Float16* __restrict__ BT,
    const float* __restrict__ bias, const float* __restrict__ resid,
    _Float16* __restrict__ out16, float* __restrict__ out32,
    int M, int N, int K) {
  __shared__ _Float16 Atile[2][128][32];

  const int tid  = threadIdx.x;
  const int wave = tid >> 5;
  const int lane = tid & 31;
  const int l16  = lane & 15;
  const int hi   = lane >> 4;
  const int wr   = wave >> 2;        // 0..1  -> 64 rows each
  const int wc   = wave & 3;         // 0..3  -> 32 cols each

  const int rowblk  = blockIdx.y * 128;
  const int rowbase = rowblk + wr * 64;
  const int colbase = blockIdx.x * 128 + wc * 32;

  v8f acc[4][2] = {};

  // stage tile t (k0 = t*32) into Atile[buf]: 512 chunks of 16B, 2 per thread
  auto issue_tile = [&](int t, int buf) {
    const int k0 = t * 32;
#pragma unroll
    for (int c = tid; c < 512; c += 256) {
      const int row = c >> 2;
      const int col = (c & 3) * 8;
      copy16_g2l(A + (size_t)(rowblk + row) * K + k0 + col, &Atile[buf][row][col]);
    }
  };

  const int nT = K / 32;
  issue_tile(0, 0);

  for (int t = 0; t < nT; ++t) {
    async_wait0();                 // this wave's DMA for tile t complete
    __syncthreads();               // tile t visible; everyone done reading t-1
    if (t + 1 < nT) issue_tile(t + 1, (t + 1) & 1);   // overlaps compute below

    const int k0 = t * 32;
    int kn = k0 + 32; if (kn == K) kn = 0;   // in-bounds prefetch target
    v16h bf[2];
#pragma unroll
    for (int j = 0; j < 2; ++j) {
      const int col = colbase + j * 16 + l16;
      bf[j] = frag_b(BT + (size_t)col * K + k0 + hi * 16);
      __builtin_prefetch(BT + (size_t)col * K + kn + hi * 16, 0, 0);
    }
#pragma unroll
    for (int i = 0; i < 4; ++i) {
      v16h af = frag_a(&Atile[t & 1][wr * 64 + i * 16 + l16][hi * 16]);
#pragma unroll
      for (int j = 0; j < 2; ++j) acc[i][j] = wmma(af, bf[j], acc[i][j]);
    }
  }

#pragma unroll
  for (int i = 0; i < 4; ++i) {
#pragma unroll
    for (int j = 0; j < 2; ++j) {
      const int col = colbase + j * 16 + l16;
      const float bv = bias[col];
#pragma unroll
      for (int r = 0; r < 8; ++r) {
        const int m = rowbase + i * 16 + r + hi * 8;
        float val = acc[i][j][r] + bv;
        if constexpr (EPI == 1)
          val = 0.5f * val * (1.0f + erff(val * 0.70710678118654752f));
        if constexpr (EPI == 2) {
          out32[(size_t)m * N + col] = resid[(size_t)m * N + col] + val;
        } else {
          out16[(size_t)m * N + col] = (_Float16)val;
        }
      }
    }
  }
}

// ---------------------------------------------------------------------------
// Scatter qkv[T][1536] (f16) -> q,k [bh][seq][64] and vT [bh][64][seq]
// ---------------------------------------------------------------------------
__global__ void k_qkv_reorder(const _Float16* __restrict__ qkv, _Float16* __restrict__ q,
                              _Float16* __restrict__ k, _Float16* __restrict__ vT) {
  size_t idx = (size_t)blockIdx.x * 256 + threadIdx.x;
  if (idx >= (size_t)TOK * H3) return;
  int t = (int)(idx / H3), c = (int)(idx % H3);
  int which = c / EMBED, rem = c % EMBED;
  int h = rem / HDIM, d = rem % HDIM;
  int b = t / SEQ, tt = t % SEQ;
  int bh = b * HEADS + h;
  _Float16 val = qkv[idx];
  if (which == 0)      q[((size_t)bh * SEQ + tt) * HDIM + d]  = val;
  else if (which == 1) k[((size_t)bh * SEQ + tt) * HDIM + d]  = val;
  else                 vT[((size_t)bh * HDIM + d) * SEQ + tt] = val;
}

// ---------------------------------------------------------------------------
// Flash attention.  block = 128 (4 waves, 1/SIMD); each wave: 16 query rows,
// streams keys in tiles of 32.  QK^T: 4 wmma; online softmax (permlane16
// row-butterflies); P via per-wave LDS transpose; P*V: 4 wmma.
// ---------------------------------------------------------------------------
__global__ __launch_bounds__(128) __attribute__((amdgpu_waves_per_eu(1)))
void k_flash_attn(const _Float16* __restrict__ q, const _Float16* __restrict__ k,
                  const _Float16* __restrict__ vT, _Float16* __restrict__ out) {
  __shared__ _Float16 plds[4][16][32];

  const int wave = threadIdx.x >> 5;
  const int lane = threadIdx.x & 31;
  const int l16  = lane & 15;
  const int hi   = lane >> 4;
  const int bh   = blockIdx.y;
  const int b    = bh / HEADS;
  const int h    = bh % HEADS;
  const int tt0  = blockIdx.x * 64 + wave * 16;
  const float scale = 0.125f;   // HDIM^-0.5

  const _Float16* qb  = q  + (size_t)bh * SEQ * HDIM;
  const _Float16* kb_ = k  + (size_t)bh * SEQ * HDIM;
  const _Float16* vb  = vT + (size_t)bh * HDIM * SEQ;

  v16h qa[2];
#pragma unroll
  for (int s = 0; s < 2; ++s)
    qa[s] = frag_a(qb + (size_t)(tt0 + l16) * HDIM + s * 32 + hi * 16);

  v8f O[4] = {};
  float m_[8], l_[8];
#pragma unroll
  for (int r = 0; r < 8; ++r) { m_[r] = -INFINITY; l_[r] = 0.f; }

  for (int kt = 0; kt < SEQ; kt += 32) {
    // ---- S = Q K^T (two 16x16 key tiles) ----
    v8f s0 = {}, s1 = {};
#pragma unroll
    for (int s = 0; s < 2; ++s) {
      v16h b0 = frag_b(kb_ + (size_t)(kt + l16)      * HDIM + s * 32 + hi * 16);
      v16h b1 = frag_b(kb_ + (size_t)(kt + 16 + l16) * HDIM + s * 32 + hi * 16);
      s0 = wmma(qa[s], b0, s0);
      s1 = wmma(qa[s], b1, s1);
    }

    // ---- online softmax: rows live in (VGPR idx r, lane-half) ----
    float alpha[8];
#pragma unroll
    for (int r = 0; r < 8; ++r) {
      float v0 = s0[r] * scale, v1 = s1[r] * scale;
      float mnew = fmaxf(m_[r], rowmax16(fmaxf(v0, v1)));
      alpha[r] = __expf(m_[r] - mnew);
      float p0 = __expf(v0 - mnew);
      float p1 = __expf(v1 - mnew);
      l_[r] = l_[r] * alpha[r] + rowsum16(p0 + p1);
      m_[r] = mnew;
      plds[wave][r + hi * 8][l16]      = (_Float16)p0;   // C-layout -> LDS
      plds[wave][r + hi * 8][16 + l16] = (_Float16)p1;
    }

    // ---- rescale O ----
#pragma unroll
    for (int t = 0; t < 4; ++t)
#pragma unroll
      for (int r = 0; r < 8; ++r) O[t][r] *= alpha[r];

    // ---- P (A-layout from LDS) x V ----
    v16h pa = frag_a(&plds[wave][l16][hi * 16]);
#pragma unroll
    for (int t = 0; t < 4; ++t) {
      v16h bv = frag_b(vb + (size_t)(t * 16 + l16) * SEQ + kt + hi * 16);
      O[t] = wmma(pa, bv, O[t]);
    }
  }

  // ---- normalize & write out[token][h*64+d] ----
#pragma unroll
  for (int r = 0; r < 8; ++r) {
    const float inv = 1.0f / l_[r];
    const int tok = b * SEQ + tt0 + r + hi * 8;
#pragma unroll
    for (int t = 0; t < 4; ++t)
      out[(size_t)tok * EMBED + h * HDIM + t * 16 + l16] = (_Float16)(O[t][r] * inv);
  }
}

// ---------------------------------------------------------------------------
extern "C" void kernel_launch(void* const* d_in, const int* in_sizes, int n_in,
                              void* d_out, int out_size, void* d_ws, size_t ws_size,
                              hipStream_t stream) {
  const float* x      = (const float*)d_in[0];
  const float* ln1_g  = (const float*)d_in[1];
  const float* ln1_b  = (const float*)d_in[2];
  const float* w_qkv  = (const float*)d_in[3];
  const float* b_qkv  = (const float*)d_in[4];
  const float* w_proj = (const float*)d_in[5];
  const float* b_proj = (const float*)d_in[6];
  const float* ln2_g  = (const float*)d_in[7];
  const float* ln2_b  = (const float*)d_in[8];
  const float* w1     = (const float*)d_in[9];
  const float* b1     = (const float*)d_in[10];
  const float* w2     = (const float*)d_in[11];
  const float* b2     = (const float*)d_in[12];
  float* out = (float*)d_out;

  char* p = (char*)d_ws;
  auto alloc16 = [&](size_t n) { _Float16* r = (_Float16*)p; p += n * sizeof(_Float16); return r; };
  auto alloc32 = [&](size_t n) { float* r = (float*)p; p += n * sizeof(float); return r; };

  _Float16* wqkvT  = alloc16((size_t)H3 * EMBED);
  _Float16* wprojT = alloc16((size_t)EMBED * EMBED);
  _Float16* w1T    = alloc16((size_t)MLPH * EMBED);
  _Float16* w2T    = alloc16((size_t)EMBED * MLPH);
  _Float16* h16    = alloc16((size_t)TOK * EMBED);
  _Float16* qkv16  = alloc16((size_t)TOK * H3);
  _Float16* q16    = alloc16((size_t)TOK * HDIM);
  _Float16* k16    = alloc16((size_t)TOK * HDIM);
  _Float16* vT16   = alloc16((size_t)TOK * HDIM);
  _Float16* attn16 = alloc16((size_t)TOK * EMBED);
  _Float16* ln2_16 = alloc16((size_t)TOK * EMBED);
  _Float16* mlp116 = alloc16((size_t)TOK * MLPH);
  float*    x2     = alloc32((size_t)TOK * EMBED);
  (void)ws_size; (void)in_sizes; (void)n_in; (void)out_size;

  // 1) weights -> f16, N-major
  k_transpose_cvt<<<(EMBED * H3   + 255) / 256, 256, 0, stream>>>(w_qkv,  wqkvT,  EMBED, H3);
  k_transpose_cvt<<<(EMBED * EMBED+ 255) / 256, 256, 0, stream>>>(w_proj, wprojT, EMBED, EMBED);
  k_transpose_cvt<<<(EMBED * MLPH + 255) / 256, 256, 0, stream>>>(w1,     w1T,    EMBED, MLPH);
  k_transpose_cvt<<<(MLPH * EMBED + 255) / 256, 256, 0, stream>>>(w2,     w2T,    MLPH,  EMBED);

  // 2) LN1
  k_layernorm_f16<<<TOK / 8, 256, 0, stream>>>(x, ln1_g, ln1_b, h16);

  // 3) QKV GEMM
  k_wmma_gemm<0><<<dim3(H3 / 128, TOK / 128), 256, 0, stream>>>(
      h16, wqkvT, b_qkv, nullptr, qkv16, nullptr, TOK, H3, EMBED);

  // 4) reorder
  k_qkv_reorder<<<(int)(((size_t)TOK * H3 + 255) / 256), 256, 0, stream>>>(qkv16, q16, k16, vT16);

  // 5) flash attention
  k_flash_attn<<<dim3(SEQ / 64, BATCH * HEADS), 128, 0, stream>>>(q16, k16, vT16, attn16);

  // 6) proj + residual -> x2
  k_wmma_gemm<2><<<dim3(EMBED / 128, TOK / 128), 256, 0, stream>>>(
      attn16, wprojT, b_proj, x, nullptr, x2, TOK, EMBED, EMBED);

  // 7) LN2
  k_layernorm_f16<<<TOK / 8, 256, 0, stream>>>(x2, ln2_g, ln2_b, ln2_16);

  // 8) MLP1 + GELU
  k_wmma_gemm<1><<<dim3(MLPH / 128, TOK / 128), 256, 0, stream>>>(
      ln2_16, w1T, b1, nullptr, mlp116, nullptr, TOK, MLPH, EMBED);

  // 9) MLP2 + residual -> out
  k_wmma_gemm<2><<<dim3(EMBED / 128, TOK / 128), 256, 0, stream>>>(
      mlp116, w2T, b2, x2, nullptr, out, TOK, EMBED, MLPH);
}
